// RNN_85950885528212
// MI455X (gfx1250) — matrix operand order repaired and checked by reading, not compiled, not run on hardware
//
#include <hip/hip_runtime.h>
#include <hip/hip_bf16.h>

#define HIDDEN 128
#define TSTEPS 512
#define FIN 2
#define KPAD 160          // 130 rounded up to 5 chunks of K=32
#define ALPHA 0.2f
#define BETA 0.7f

typedef _Float16 h8   __attribute__((ext_vector_type(8)));    // 4 VGPRs
typedef _Float16 v16h __attribute__((ext_vector_type(16)));   // 8 VGPRs (WMMA A/B)
typedef float    v8f  __attribute__((ext_vector_type(8)));    // 8 VGPRs (WMMA C/D)

static __device__ __forceinline__ uint32_t pack2h(float a, float b) {
    union { _Float16 h[2]; uint32_t u; } q;
    q.h[0] = (_Float16)a; q.h[1] = (_Float16)b;
    return q.u;
}

static __device__ __forceinline__ float fast_tanh(float z) {
#if __has_builtin(__builtin_amdgcn_tanhf)
    return __builtin_amdgcn_tanhf(z);   // v_tanh_f32 on gfx1250
#else
    return tanhf(z);
#endif
}

// One block of 256 threads (8 wave32) owns 16 batch rows for the whole scan.
// Wave w owns hidden columns [16w, 16w+16): its W_rec B-tiles stay in VGPRs.
// conc is ping-pong double buffered: step t reads buf[t&1], writes buf[(t+1)&1]
// -> exactly one barrier per timestep.
__global__ __launch_bounds__(256) void RNN_85950885528212_kernel(
    const float* __restrict__ x,     // (B, 512, 2)
    const float* __restrict__ Wrec,  // (130, 128) rows 0..1 = x-weights, 2..129 = h-weights
    const float* __restrict__ brec,  // (128,)
    const float* __restrict__ Wout,  // (128, 1)
    const float* __restrict__ bout,  // (1,)
    float* __restrict__ out)         // (B,)
{
    // conc row m (f16, stride 160): cols 0..127 = h, 128..129 = x_t, 130..159 = 0
    __shared__ __align__(16) _Float16 conc[2][16 * KPAD];       // 2 x 5 KiB
    __shared__ uint32_t xbuf[16 * TSTEPS];                      // 32 KiB packed (x0,x1) f16
    __shared__ float    sig2buf[16];

    const int tid  = threadIdx.x;
    const int wave = tid >> 5;
    const int lane = tid & 31;
    const int lhi  = lane >> 4;      // half-wave select
    const int lm   = lane & 15;
    const int ncol = wave * 16 + lm; // this lane's hidden column
    const int b0   = blockIdx.x * 16;

    // ---- prologue: coalesced copy of the 16-row x slab into LDS (f32 -> packed f16)
    {
        const float4* gx = (const float4*)(x + (size_t)b0 * (TSTEPS * FIN));
        #pragma unroll
        for (int i = 0; i < 16; ++i) {
            int idx = tid + i * 256;           // 4096 float4 total
            float4 v = gx[idx];
            int fl = idx * 4;
            int m  = fl >> 10;                 // / (512*2)
            int t0 = (fl & 1023) >> 1;
            xbuf[m * TSTEPS + t0]     = pack2h(v.x, v.y);
            xbuf[m * TSTEPS + t0 + 1] = pack2h(v.z, v.w);
        }
    }
    { // zero both conc buffers (h=0 initial state; pad columns stay zero forever)
        uint32_t* cz = (uint32_t*)&conc[0][0];
        #pragma unroll
        for (int i = 0; i < 10; ++i) cz[tid + 256 * i] = 0u;
    }
    if (tid < 16) sig2buf[tid] = 0.0f;

    // ---- stage W_rec into WMMA B-layout registers: 5 K-chunks x 8 VGPRs (packed f16)
    // B k order (contiguous, A same): lanes 0-15 K=0..15 (VGPR v holds K=2v,2v+1), lanes 16-31 K=16..31
    // conc K remap: k<128 -> W_rec row k+2 (h part); k=128,129 -> rows 0,1 (x part); else 0
    v16h B[5];
    #pragma unroll
    for (int c = 0; c < 5; ++c) {
        #pragma unroll
        for (int v = 0; v < 8; ++v) {
            int k0 = c * 32 + lhi * 16 + 2 * v;
            int k1 = k0 + 1;
            int r0 = (k0 < 128) ? (k0 + 2) : (k0 < 130 ? (k0 - 128) : -1);
            int r1 = (k1 < 128) ? (k1 + 2) : (k1 < 130 ? (k1 - 128) : -1);
            float f0 = (r0 >= 0) ? Wrec[r0 * HIDDEN + ncol] : 0.0f;
            float f1 = (r1 >= 0) ? Wrec[r1 * HIDDEN + ncol] : 0.0f;
            B[c][2 * v]     = (_Float16)f0;
            B[c][2 * v + 1] = (_Float16)f1;
        }
    }
    const float bias = brec[ncol];
    const float wo   = Wout[ncol];
    const float bo   = bout[0];

    // sig2 is linear in omega: each wave accumulates beta-discounted partial sums
    float acc[8];
    #pragma unroll
    for (int v = 0; v < 8; ++v) acc[v] = 0.0f;
    float accR = 0.0f, resid = 0.0f;   // alpha*resid^2 + b_out stream (lanes tid<16)

    for (int t = 0; t < TSTEPS; ++t) {
        _Float16* bufR = &conc[t & 1][0];         // read  [h_{t-1} | x_t | 0]
        _Float16* bufW = &conc[(t + 1) & 1][0];   // write h_t

        if (tid < 16) {   // lane m = tid: drop x_t into read-buffer cols 128..129
            uint32_t xp = xbuf[tid * TSTEPS + t];
            *(uint32_t*)(bufR + tid * KPAD + 128) = xp;
            union { uint32_t u; _Float16 h[2]; } q; q.u = xp;
            accR  = accR * BETA + (resid * resid * ALPHA + bo);
            resid = (float)q.h[0];
        }
        __syncthreads();   // the ONLY barrier: writes into bufR visible before reads

        // issue all 10 A-chunk loads into distinct registers, THEN the WMMA chain
        h8 alo[5], ahi[5];
        #pragma unroll
        for (int c = 0; c < 5; ++c) {
            const _Float16* ap = bufR + lm * KPAD + c * 32 + lhi * 16;
            alo[c] = *(const h8*)(ap);        // K = 32c + 16*lhi     .. +7
            ahi[c] = *(const h8*)(ap + 8);    // K = 32c + 16*lhi + 8 .. +15
        }
        v8f cacc = {};
        #pragma unroll
        for (int c = 0; c < 5; ++c) {
            v16h a = __builtin_shufflevector(alo[c], ahi[c],
                        0,1,2,3,4,5,6,7,8,9,10,11,12,13,14,15);
            cacc = __builtin_amdgcn_wmma_f32_16x16x32_f16(
                        false, a, false, B[c], (short)0, cacc, false, false);
        }

        // bias + tanh + beta-discounted omega partials (co-executes with XDL),
        // then scatter h_t into the WRITE buffer (read at step t+1, after its barrier)
        #pragma unroll
        for (int v = 0; v < 8; ++v) {
            float th = fast_tanh(cacc[v] + bias);
            acc[v] = acc[v] * BETA + th * wo;
            int m = v + lhi * 8;                 // C layout: rows m = v + 8*lhi
            bufW[m * KPAD + ncol] = (_Float16)th;
        }

#if __has_builtin(__builtin_amdgcn_sched_group_barrier)
        // Pin the schedule: 10 DS reads first (distinct regs, pipelined dscnt waits),
        // then the 5-WMMA chain, then the VALU tail, then the 8 DS writes.
        __builtin_amdgcn_sched_group_barrier(0x100, 10, 0); // DS read  x10
        __builtin_amdgcn_sched_group_barrier(0x008,  5, 0); // MFMA/WMMA x5
        __builtin_amdgcn_sched_group_barrier(0x002, 40, 0); // VALU tail
        __builtin_amdgcn_sched_group_barrier(0x200,  8, 0); // DS write x8
#endif
    }

    // ---- final reduction over hidden columns (16 lanes) and waves (8)
    #pragma unroll
    for (int v = 0; v < 8; ++v) {
        int m = v + lhi * 8;
        atomicAdd(&sig2buf[m], acc[v]);    // ds_add_f32
    }
    if (tid < 16) atomicAdd(&sig2buf[tid], accR);
    __syncthreads();
    if (tid < 16) out[b0 + tid] = sig2buf[tid];
}

extern "C" void kernel_launch(void* const* d_in, const int* in_sizes, int n_in,
                              void* d_out, int out_size, void* d_ws, size_t ws_size,
                              hipStream_t stream) {
    const float* x    = (const float*)d_in[0];  // input_tensor (B,512,2)
    const float* Wrec = (const float*)d_in[1];  // (130,128)
    const float* brec = (const float*)d_in[2];  // (128,)
    const float* Wout = (const float*)d_in[3];  // (128,1)
    const float* bout = (const float*)d_in[4];  // (1,)
    float* out = (float*)d_out;

    int rows   = in_sizes[0] / (TSTEPS * FIN);  // B = 8192
    int blocks = rows / 16;                     // 512 blocks x 256 threads
    RNN_85950885528212_kernel<<<blocks, 256, 0, stream>>>(x, Wrec, brec, Wout, bout, out);
}
